// MultiAggLP_18159121728000
// MI455X (gfx1250) — matrix operand order repaired and checked by reading, not compiled, not run on hardware
//
#include <hip/hip_runtime.h>
#include <math.h>

typedef __attribute__((ext_vector_type(16))) _Float16 v16h;
typedef __attribute__((ext_vector_type(8)))  _Float16 v8h;
typedef __attribute__((ext_vector_type(8)))  float    v8f;

#define TT   8
#define NNODE 8192
#define EDG  131072
#define NH   8
#define DHD  16
#define NC   100
#define DG   128
#define DAGG 384
#define DR   256

#define BM 128
#define BN 64
#define BK 32
#define LP 8          // LDS row pad (stride BK+LP = 40 f16 = 80B, 16B-aligned rows)

__device__ __forceinline__ float sigmoidf_(float x) { return 1.0f / (1.0f + expf(-x)); }

// float atomic max via sign-split int/uint atomics (correct for non-NaN IEEE)
__device__ __forceinline__ void atomicMaxF(float* addr, float val) {
  if (val >= 0.0f) atomicMax((int*)addr, __float_as_int(val));
  else             atomicMin((unsigned int*)addr, __float_as_uint(val));
}

// ---------------------------------------------------------------------------
// Block-tiled WMMA GEMM: C[M,Ncols] = act( init + A[M,K] @ B[K,Ncols] )
//   init: existing C (acc=1), bias[col] (bias!=null), else 0.
//   Block = 256 threads (8 waves). Tile BM=128 x BN=64, K-step 32.
//   As is [m][k]; Bs is stored TRANSPOSED [n][k] so each lane's fragment is
//   two contiguous 16B chunks -> ds_load_b128 (no scalar u16 gather).
//   Requires: M % 128 == 0, Ncols % 64 == 0, K % 32 == 0 (true for all uses).
// ---------------------------------------------------------------------------
__global__ void wmma_gemm_f16(const float* __restrict__ A, const float* __restrict__ B,
                              const float* __restrict__ bias, float* __restrict__ Cm,
                              int M, int Ncols, int K, int acc, int act)
{
  __shared__ _Float16 As[BM][BK + LP];
  __shared__ _Float16 Bs[BN][BK + LP];   // transposed: [n][k]

  int tid  = threadIdx.x;
  int lane = tid & 31;
  int w    = tid >> 5;                       // wave 0..7
  int nTiles = Ncols / BN;
  int bm = (blockIdx.x / nTiles) * BM;
  int bn = (blockIdx.x % nTiles) * BN;

  int lr   = lane & 15;
  int half = lane >> 4;
  int row0 = bm + w * 16 + half * 8;         // C fragment rows row0..row0+7
  int kg8  = half * 8, kg16 = half * 16;

  v8f accv[4];
#pragma unroll
  for (int nn = 0; nn < 4; ++nn) {
    int col = bn + nn * 16 + lr;
    if (acc) {
#pragma unroll
      for (int r = 0; r < 8; ++r) accv[nn][r] = Cm[(size_t)(row0 + r) * Ncols + col];
    } else if (bias) {
      float bv = bias[col];
#pragma unroll
      for (int r = 0; r < 8; ++r) accv[nn][r] = bv;
    } else {
#pragma unroll
      for (int r = 0; r < 8; ++r) accv[nn][r] = 0.0f;
    }
  }

  for (int k0 = 0; k0 < K; k0 += BK) {
    // stage A tile: 128x32 f32 -> f16 LDS; coalesced float4 reads, packed 8B LDS stores
#pragma unroll
    for (int it = 0; it < 4; ++it) {
      int lin = tid + it * 256;
      int r  = lin >> 3;                     // BK/4 = 8 float4 per row
      int c4 = (lin & 7) << 2;
      const float4 v = *(const float4*)(A + (size_t)(bm + r) * K + k0 + c4);
      union { _Float16 h[4]; float2 f; } u;
      u.h[0] = (_Float16)v.x; u.h[1] = (_Float16)v.y;
      u.h[2] = (_Float16)v.z; u.h[3] = (_Float16)v.w;
      *(float2*)&As[r][c4] = u.f;
    }
    // stage B tile transposed: coalesced float4 read of row k, scatter to Bs[n][k]
#pragma unroll
    for (int it = 0; it < 2; ++it) {
      int lin = tid + it * 256;
      int r  = lin >> 4;                     // BN/4 = 16 float4 per row
      int c4 = (lin & 15) << 2;
      const float4 v = *(const float4*)(B + (size_t)(k0 + r) * Ncols + bn + c4);
      Bs[c4 + 0][r] = (_Float16)v.x;
      Bs[c4 + 1][r] = (_Float16)v.y;
      Bs[c4 + 2][r] = (_Float16)v.z;
      Bs[c4 + 3][r] = (_Float16)v.w;
    }
    __syncthreads();

    const _Float16* ap = &As[w * 16 + lr][kg8];
    v8h a0 = *(const v8h*)ap;                // cols kg8..kg8+7      (16B aligned)
    v8h a1 = *(const v8h*)(ap + 16);         // cols 16+kg8..16+kg8+7
    v16h a = __builtin_shufflevector(a0, a1, 0, 1, 2, 3, 4, 5, 6, 7,
                                             8, 9, 10, 11, 12, 13, 14, 15);
#pragma unroll
    for (int nn = 0; nn < 4; ++nn) {
      const _Float16* bp = &Bs[nn * 16 + lr][kg8];
      v8h b0 = *(const v8h*)bp;              // K = kg8..kg8+7
      v8h b1 = *(const v8h*)(bp + 16);       // K = 16+kg8..16+kg8+7
      v16h b = __builtin_shufflevector(b0, b1, 0, 1, 2, 3, 4, 5, 6, 7,
                                               8, 9, 10, 11, 12, 13, 14, 15);
      accv[nn] = __builtin_amdgcn_wmma_f32_16x16x32_f16(false, a, false, b, (short)0,
                                                        accv[nn], false, false);
    }
    __syncthreads();
  }

#pragma unroll
  for (int nn = 0; nn < 4; ++nn) {
    int col = bn + nn * 16 + lr;
#pragma unroll
    for (int r = 0; r < 8; ++r) {
      float v = accv[nn][r];
      if (act == 1) v = tanhf(v);
      else if (act == 2) v = sigmoidf_(v);
      Cm[(size_t)(row0 + r) * Ncols + col] = v;
    }
  }
}

// ---------------------------------------------------------------------------
// Fused final output: two Gram matrices (embn, scal) via WMMA over K=64,
// epilogue adj[i,j] = 1 + tanh(-(sq[i]+sq[j]-2*G[i,j]) * S[i,j]).
// Store-bandwidth bound (268 MB of stores): one 4B store per output element;
// embn/scal (4 MB total) live in L2 across the whole launch.
// ---------------------------------------------------------------------------
__global__ void adj_kernel(const float* __restrict__ embn, const float* __restrict__ scal,
                           const float* __restrict__ sq, float* __restrict__ out)
{
  int lane = threadIdx.x & 31;
  int wave = blockIdx.x * (blockDim.x >> 5) + (threadIdx.x >> 5);
  const int tiles = NNODE >> 4;
  int ti = (wave / tiles) << 4;
  int tj = (wave % tiles) << 4;

  v8f g = {}, s = {};
  int arow = ti + (lane & 15);
  int brow = tj + (lane & 15);
  int kg8  = (lane >> 4) << 3;
  int kg16 = (lane >> 4) << 4;
#pragma unroll
  for (int k0 = 0; k0 < 64; k0 += 32) {
    v16h a, b;
    const float* Ap = embn + (size_t)arow * 64 + k0;
    const float* Bp = embn + (size_t)brow * 64 + k0;
#pragma unroll
    for (int j = 0; j < 16; ++j) a[j] = (_Float16)Ap[((j >> 3) << 4) + kg8 + (j & 7)];
#pragma unroll
    for (int j = 0; j < 16; ++j) b[j] = (_Float16)Bp[((j >> 3) << 4) + kg8 + (j & 7)];
    g = __builtin_amdgcn_wmma_f32_16x16x32_f16(false, a, false, b, (short)0, g, false, false);

    const float* Ap2 = scal + (size_t)arow * 64 + k0;
    const float* Bp2 = scal + (size_t)brow * 64 + k0;
#pragma unroll
    for (int j = 0; j < 16; ++j) a[j] = (_Float16)Ap2[((j >> 3) << 4) + kg8 + (j & 7)];
#pragma unroll
    for (int j = 0; j < 16; ++j) b[j] = (_Float16)Bp2[((j >> 3) << 4) + kg8 + (j & 7)];
    s = __builtin_amdgcn_wmma_f32_16x16x32_f16(false, a, false, b, (short)0, s, false, false);
  }

  int i0 = ti + ((lane >> 4) << 3);
  int jc = tj + (lane & 15);
  float sqj = sq[jc];
#pragma unroll
  for (int r = 0; r < 8; ++r) {
    int i = i0 + r;
    float dist = sq[i] + sqj - 2.0f * g[r];
    out[(size_t)i * NNODE + jc] = 1.0f + tanhf(-dist * s[r]);
  }
}

// ------------------------------- utility -----------------------------------
__global__ void fill_kernel(float* p, int n, float v) {
  int i = blockIdx.x * blockDim.x + threadIdx.x;
  if (i < n) p[i] = v;
}

// ------------------------------- GAT ---------------------------------------
__global__ void gat_init(float* outacc, float* m, float* den) {
  int i = blockIdx.x * blockDim.x + threadIdx.x;
  if (i < NNODE * DG) outacc[i] = 0.0f;
  if (i < NNODE * NH) { m[i] = -INFINITY; den[i] = 0.0f; }
}

__global__ void gat_scores(const float* __restrict__ h, const float* __restrict__ a_s,
                           const float* __restrict__ a_d, float* __restrict__ ssrc,
                           float* __restrict__ sdst)
{
  int idx = blockIdx.x * blockDim.x + threadIdx.x;
  if (idx >= NNODE * NH) return;
  int n = idx >> 3, hd = idx & 7;
  const float* hp = h + (size_t)n * DG + hd * DHD;
  const float* ap = a_s + hd * DHD;
  const float* dp = a_d + hd * DHD;
  float as = 0.0f, ad = 0.0f;
#pragma unroll
  for (int k = 0; k < DHD; ++k) { as += hp[k] * ap[k]; ad += hp[k] * dp[k]; }
  ssrc[idx] = as; sdst[idx] = ad;
}

__global__ void gat_edge1(const int* __restrict__ src, const int* __restrict__ dst,
                          const float* __restrict__ ew, const float* __restrict__ ssrc,
                          const float* __restrict__ sdst, float* __restrict__ ebuf,
                          float* __restrict__ m)
{
  int idx = blockIdx.x * blockDim.x + threadIdx.x;
  if (idx >= EDG * NH) return;
  int e = idx >> 3, hd = idx & 7;
  int sN = src[e], dN = dst[e];
  float v = ssrc[sN * NH + hd] + sdst[dN * NH + hd];
  v = (v > 0.0f) ? v : 0.2f * v;
  v *= ew[e];
  ebuf[idx] = v;
  atomicMaxF(&m[dN * NH + hd], v);
}

__global__ void gat_edge2(const int* __restrict__ dst, float* __restrict__ ebuf,
                          const float* __restrict__ m, float* __restrict__ den)
{
  int idx = blockIdx.x * blockDim.x + threadIdx.x;
  if (idx >= EDG * NH) return;
  int e = idx >> 3, hd = idx & 7;
  int dN = dst[e];
  float ex = expf(ebuf[idx] - m[dN * NH + hd]);
  ebuf[idx] = ex;
  atomicAdd(&den[dN * NH + hd], ex);
}

// 4 features per thread: float4 loads of h, 4 atomic f32 adds
__global__ void gat_edge3(const int* __restrict__ src, const int* __restrict__ dst,
                          const float* __restrict__ ebuf, const float* __restrict__ den,
                          const float* __restrict__ h, float* __restrict__ outacc)
{
  int idx = blockIdx.x * blockDim.x + threadIdx.x;
  if (idx >= EDG * 32) return;
  int e = idx >> 5;
  int dd = (idx & 31) << 2;                  // 0,4,...,124 (stays inside one head of 16)
  int hd = dd >> 4;
  int sN = src[e], dN = dst[e];
  float alpha = ebuf[e * NH + hd] / (den[dN * NH + hd] + 1e-16f);
  const float4 hv = *(const float4*)(h + (size_t)sN * DG + dd);
  float* op = outacc + (size_t)dN * DG + dd;
  atomicAdd(op + 0, alpha * hv.x);
  atomicAdd(op + 1, alpha * hv.y);
  atomicAdd(op + 2, alpha * hv.z);
  atomicAdd(op + 3, alpha * hv.w);
}

__global__ void gat_elu(const float* __restrict__ acc, float* __restrict__ xout) {
  int i = blockIdx.x * blockDim.x + threadIdx.x;
  if (i >= NNODE * DG) return;
  float v = acc[i];
  xout[i] = (v > 0.0f) ? v : expm1f(v);
}

// ------------------------------- fusion ------------------------------------
__global__ void fuse_pool(const float* __restrict__ xt, const int* __restrict__ com,
                          const float* __restrict__ nw, float* __restrict__ num,
                          float* __restrict__ denc, float* __restrict__ macro,
                          float* __restrict__ wsum)
{
  int idx = blockIdx.x * blockDim.x + threadIdx.x;
  if (idx >= NNODE * DG) return;
  int n = idx >> 7, dd = idx & 127;
  float w = nw[n];
  float xv = xt[idx];
  int cm = com[n];
  atomicAdd(&num[cm * DG + dd], w * xv);
  atomicAdd(&macro[dd], w * xv);
  if (dd == 0) { atomicAdd(&denc[cm], w); atomicAdd(wsum, w); }
}

__global__ void fuse_combine(const float* __restrict__ xt, const int* __restrict__ com,
                             const float* __restrict__ num, const float* __restrict__ denc,
                             const float* __restrict__ macro, const float* __restrict__ wsum,
                             float* __restrict__ agg)
{
  int n = blockIdx.x * blockDim.x + threadIdx.x;
  if (n >= NNODE) return;
  int cm = com[n];
  float invd = 1.0f / (denc[cm] + 1e-16f);
  float invw = 1.0f / (wsum[0] + 1e-16f);
  const float* xr = xt + (size_t)n * DG;
  const float* nr = num + (size_t)cm * DG;
  float m0 = 0.0f, m1 = 0.0f, m2 = 0.0f;
  for (int d = 0; d < DG; ++d) { m0 += xr[d]; m1 += nr[d] * invd; m2 += macro[d] * invw; }
  m0 *= (1.0f / DG); m1 *= (1.0f / DG); m2 *= (1.0f / DG);
  m0 = (m0 > 0.0f) ? m0 : 0.2f * m0;
  m1 = (m1 > 0.0f) ? m1 : 0.2f * m1;
  m2 = (m2 > 0.0f) ? m2 : 0.2f * m2;
  float mx = fmaxf(m0, fmaxf(m1, m2));
  float x0 = expf(m0 - mx), x1 = expf(m1 - mx), x2 = expf(m2 - mx);
  float inv = 1.0f / (x0 + x1 + x2);
  float a0 = x0 * inv, a1 = x1 * inv, a2 = x2 * inv;
  float* ar = agg + (size_t)n * DAGG;
  for (int d = 0; d < DG; ++d) {
    ar[d]          = a0 * xr[d];
    ar[DG + d]     = a1 * nr[d] * invd;
    ar[2 * DG + d] = a2 * macro[d] * invw;
  }
}

// ------------------------------- LSTM --------------------------------------
__global__ void lstm_gates(const float* __restrict__ Z, float* __restrict__ hs,
                           float* __restrict__ cs)
{
  int idx = blockIdx.x * blockDim.x + threadIdx.x;
  if (idx >= NNODE * DR) return;
  int n = idx >> 8, j = idx & 255;
  const float* z = Z + (size_t)n * 1024;
  float ig = sigmoidf_(z[j]);
  float fg = sigmoidf_(z[256 + j]);
  float gg = tanhf(z[512 + j]);
  float og = sigmoidf_(z[768 + j]);
  float c = fg * cs[idx] + ig * gg;
  cs[idx] = c;
  hs[idx] = og * tanhf(c);
}

// ----------------------------- epilogue prep -------------------------------
__global__ void colnorm_kernel(const float* __restrict__ emb, float* __restrict__ norm2) {
  __shared__ float red[256];
  int d = blockIdx.x;
  float acc = 0.0f;
  for (int n = threadIdx.x; n < NNODE; n += blockDim.x) {
    float v = emb[(size_t)n * 64 + d];
    acc += v * v;
  }
  red[threadIdx.x] = acc;
  __syncthreads();
  for (int s = 128; s > 0; s >>= 1) {
    if (threadIdx.x < s) red[threadIdx.x] += red[threadIdx.x + s];
    __syncthreads();
  }
  if (threadIdx.x == 0) norm2[d] = fmaxf(sqrtf(red[0]), 1e-12f);
}

__global__ void embn_kernel(const float* __restrict__ emb, const float* __restrict__ norm2,
                            float* __restrict__ embn) {
  int i = blockIdx.x * blockDim.x + threadIdx.x;
  if (i < NNODE * 64) embn[i] = emb[i] / norm2[i & 63];
}

__global__ void sq_kernel(const float* __restrict__ embn, float* __restrict__ sq) {
  int n = blockIdx.x * blockDim.x + threadIdx.x;
  if (n >= NNODE) return;
  const float* r = embn + (size_t)n * 64;
  float a = 0.0f;
#pragma unroll
  for (int d = 0; d < 64; ++d) a += r[d] * r[d];
  sq[n] = a;
}

// ===========================================================================
extern "C" void kernel_launch(void* const* d_in, const int* in_sizes, int n_in,
                              void* d_out, int out_size, void* d_ws, size_t ws_size,
                              hipStream_t stream) {
  const int*   edge_index = (const int*)d_in[0];
  const float* edge_weight= (const float*)d_in[1];
  const float* feat       = (const float*)d_in[2];
  const int*   com_ids    = (const int*)d_in[3];
  const float* node_w     = (const float*)d_in[4];
  const float* gat_W      = (const float*)d_in[5];
  const float* gat_asrc   = (const float*)d_in[6];
  const float* gat_adst   = (const float*)d_in[7];
  const float* lstm_Wx    = (const float*)d_in[8];
  const float* lstm_Wh    = (const float*)d_in[9];
  const float* lstm_b     = (const float*)d_in[10];
  const float* emb_W0     = (const float*)d_in[11];
  const float* emb_b0     = (const float*)d_in[12];
  const float* emb_W1     = (const float*)d_in[13];
  const float* emb_b1     = (const float*)d_in[14];
  const float* scal_W0    = (const float*)d_in[15];
  const float* scal_b0    = (const float*)d_in[16];
  const float* scal_W1    = (const float*)d_in[17];
  const float* scal_b1    = (const float*)d_in[18];
  float* out = (float*)d_out;

  // ---- workspace carve (floats, all offsets 16B-aligned); Z aliases xB ----
  float* p = (float*)d_ws;
  float* xB    = p; p += (size_t)TT * NNODE * DG;   // GAT layer-0 out / LSTM Z (N*1024)
  float* xA    = p; p += (size_t)TT * NNODE * DG;   // GAT layer-1 out
  float* hbuf  = p; p += (size_t)NNODE * DG;
  float* ssrc  = p; p += NNODE * NH;
  float* sdst  = p; p += NNODE * NH;
  float* ebuf  = p; p += (size_t)EDG * NH;
  float* mbuf  = p; p += NNODE * NH;
  float* dena  = p; p += NNODE * NH;
  float* outac = p; p += (size_t)NNODE * DG;
  float* aggt  = p; p += (size_t)NNODE * DAGG;
  float* hs    = p; p += (size_t)NNODE * DR;
  float* cs    = p; p += (size_t)NNODE * DR;
  float* num   = p; p += NC * DG;
  float* denc  = p; p += NC;
  float* macro = p; p += DG;
  float* wsum  = p; p += 4;                         // padded for 16B alignment
  float* e1    = p; p += (size_t)NNODE * 128;
  float* s1    = p; p += (size_t)NNODE * 128;
  float* scal  = p; p += (size_t)NNODE * 64;
  float* embn  = p; p += (size_t)NNODE * 64;
  float* norm2 = p; p += 64;
  float* sq    = p; p += NNODE;
  float* Z     = xB;

  dim3 blk(256);
  auto gblk = [](int M, int N) { return (M / BM) * (N / BN); };

  // ================= GAT: 2 layers x T snapshots =================
  for (int l = 0; l < 2; ++l) {
    const float* Wl  = gat_W + (size_t)l * DG * DG;
    const float* asl = gat_asrc + l * NH * DHD;
    const float* adl = gat_adst + l * NH * DHD;
    const float* xin_base  = (l == 0) ? feat : xB;
    float*       xout_base = (l == 0) ? xB : xA;
    for (int t = 0; t < TT; ++t) {
      const float* xin  = xin_base + (size_t)t * NNODE * DG;
      float*       xout = xout_base + (size_t)t * NNODE * DG;
      const int*   src  = edge_index + (size_t)t * 2 * EDG;
      const int*   dst  = src + EDG;
      const float* ew   = edge_weight + (size_t)t * EDG;

      wmma_gemm_f16<<<gblk(NNODE, DG), 256, 0, stream>>>(xin, Wl, nullptr, hbuf,
                                                         NNODE, DG, DG, 0, 0);
      gat_scores<<<(NNODE * NH + 255) / 256, blk, 0, stream>>>(hbuf, asl, adl, ssrc, sdst);
      gat_init<<<(NNODE * DG + 255) / 256, blk, 0, stream>>>(outac, mbuf, dena);
      gat_edge1<<<(EDG * NH + 255) / 256, blk, 0, stream>>>(src, dst, ew, ssrc, sdst, ebuf, mbuf);
      gat_edge2<<<(EDG * NH + 255) / 256, blk, 0, stream>>>(dst, ebuf, mbuf, dena);
      gat_edge3<<<(EDG * 32 + 255) / 256, blk, 0, stream>>>(src, dst, ebuf, dena, hbuf, outac);
      gat_elu<<<(NNODE * DG + 255) / 256, blk, 0, stream>>>(outac, xout);
    }
  }

  // ================= fusion + LSTM (sequential over T) =================
  fill_kernel<<<(2 * NNODE * DR + 255) / 256, blk, 0, stream>>>(hs, 2 * NNODE * DR, 0.0f);
  for (int t = 0; t < TT; ++t) {
    const float* xt = xA + (size_t)t * NNODE * DG;
    int poolN = NC * DG + NC + DG + 4;   // num,denc,macro,wsum contiguous
    fill_kernel<<<(poolN + 255) / 256, blk, 0, stream>>>(num, poolN, 0.0f);
    fuse_pool<<<(NNODE * DG + 255) / 256, blk, 0, stream>>>(
        xt, com_ids + (size_t)t * NNODE, node_w + (size_t)t * NNODE, num, denc, macro, wsum);
    fuse_combine<<<(NNODE + 255) / 256, blk, 0, stream>>>(
        xt, com_ids + (size_t)t * NNODE, num, denc, macro, wsum, aggt);

    wmma_gemm_f16<<<gblk(NNODE, 1024), 256, 0, stream>>>(aggt, lstm_Wx, lstm_b, Z,
                                                         NNODE, 1024, DAGG, 0, 0);
    wmma_gemm_f16<<<gblk(NNODE, 1024), 256, 0, stream>>>(hs, lstm_Wh, nullptr, Z,
                                                         NNODE, 1024, DR, 1, 0);
    lstm_gates<<<(NNODE * DR + 255) / 256, blk, 0, stream>>>(Z, hs, cs);
  }

  // ================= decoders =================
  float* emb_out = out + (size_t)NNODE * NNODE;   // second tuple output
  wmma_gemm_f16<<<gblk(NNODE, 128), 256, 0, stream>>>(hs, emb_W0, emb_b0, e1,
                                                      NNODE, 128, DR, 0, 1);
  wmma_gemm_f16<<<gblk(NNODE, 64), 256, 0, stream>>>(e1, emb_W1, emb_b1, emb_out,
                                                     NNODE, 64, 128, 0, 1);
  wmma_gemm_f16<<<gblk(NNODE, 128), 256, 0, stream>>>(hs, scal_W0, scal_b0, s1,
                                                      NNODE, 128, DR, 0, 2);
  wmma_gemm_f16<<<gblk(NNODE, 64), 256, 0, stream>>>(s1, scal_W1, scal_b1, scal,
                                                     NNODE, 64, 128, 0, 2);

  // ================= epilogue: normalize + fused N x N output =================
  colnorm_kernel<<<64, 256, 0, stream>>>(emb_out, norm2);
  embn_kernel<<<(NNODE * 64 + 255) / 256, blk, 0, stream>>>(emb_out, norm2, embn);
  sq_kernel<<<(NNODE + 255) / 256, blk, 0, stream>>>(embn, sq);
  int wadj = (NNODE / 16) * (NNODE / 16);  // 262144 waves, 8 per block
  adj_kernel<<<wadj / 8, 256, 0, stream>>>(embn, scal, sq, out);
}
